// StockLSTM_55551107006992
// MI455X (gfx1250) — matrix operand order repaired and checked by reading, not compiled, not run on hardware
//
#include <hip/hip_runtime.h>
#include <hip/hip_bf16.h>

// ---------------- problem constants ----------------
#define IN_DIM 801
#define KPAD   832            // 801 padded to multiple of 32
#define T_DIM  512
#define B_DIM  128
#define H_DIM  64
#define G_DIM  256            // 4*H

typedef __attribute__((ext_vector_type(16))) __bf16 v16bf;
typedef __attribute__((ext_vector_type(8)))  float  v8f;
typedef unsigned int v4u  __attribute__((ext_vector_type(4)));
typedef int          v8i32 __attribute__((ext_vector_type(8)));
typedef int          v4i32 __attribute__((ext_vector_type(4)));

#if defined(__has_builtin)
#if __has_builtin(__builtin_amdgcn_tensor_load_to_lds) && \
    __has_builtin(__builtin_amdgcn_s_wait_tensorcnt)
#define HAVE_TDM 1
#endif
#endif

// ---------------- helpers ----------------
__device__ __forceinline__ unsigned short f2bf_bits(float f) {
  unsigned u = __builtin_bit_cast(unsigned, f);
  u += 0x7FFFu + ((u >> 16) & 1u);               // round-to-nearest-even
  return (unsigned short)(u >> 16);
}
__device__ __forceinline__ __bf16 f2bf(float f) {
  return __builtin_bit_cast(__bf16, f2bf_bits(f));
}
__device__ __forceinline__ float sigmoidf_(float x) {
  return 1.0f / (1.0f + __expf(-x));
}
__device__ __forceinline__ v8f v8f_zero() {
  v8f z;
#pragma unroll
  for (int e = 0; e < 8; ++e) z[e] = 0.0f;
  return z;
}

// ISA 7.12.2: 16-bit A matrix 16x32 (MxK), wave32.
__device__ __forceinline__ void a_frag_pos(int m, int k, int& lane, int& elem) {
  int p = k & 1;
  int q = k >> 1;                 // 0..15
  int half = (q >> 2) & 1;        // K block of 8
  int v = (q & 3) + ((q >> 3) << 2);
  lane = m + 16 * half;
  elem = 2 * v + p;
}
// ISA 7.12.2: 16-bit B matrix 32x16 (KxN), wave32.
__device__ __forceinline__ void b_frag_pos(int k, int n, int& lane, int& elem) {
  lane = (((k >> 3) & 1) << 4) + n;
  elem = (k & 7) + ((k >> 4) << 3);
}

// ---- Tensor Data Mover: 1-D contiguous load of `ndw` DWORDs into LDS ----
// D# built per ISA 8.3/8.4: group0 {count=1, lds_addr, global_addr, type=2},
// group1 {data_size=4B, tensor_dim0=tile_dim0=ndw, tile_dim1=tensor_dim1=1}.
// 6-arg builtin on this toolchain: (g0, g1, g2, g3, g4, cpol).
__device__ __forceinline__ void tdm_load_dwords(void* ldsDst, const void* gsrc,
                                                unsigned ndw) {
#ifdef HAVE_TDM
  unsigned long ga = (unsigned long)gsrc;
  unsigned lds_off = (unsigned)(size_t)ldsDst;   // low 32 bits = LDS byte offset
  v4u g0;
  g0[0] = 1u;                                        // count=1, user D#
  g0[1] = lds_off;                                   // lds_addr
  g0[2] = (unsigned)(ga & 0xFFFFFFFFu);              // global_addr[31:0]
  g0[3] = (unsigned)((ga >> 32) & 0x1FFFFFFu) | (2u << 30);  // [56:32], type=2
  v8i32 g1;
  g1[0] = (int)(2u << 16);            // wg_mask=0, data_size=2 (4 bytes)
  g1[1] = (int)(ndw << 16);           // tensor_dim0[15:0] @ bits 63:48
  g1[2] = (int)((ndw >> 16) | (1u << 16)); // tensor_dim0[31:16], tensor_dim1=1
  g1[3] = (int)((ndw & 0xFFFFu) << 16);    // tile_dim0 @ bits 127:112
  g1[4] = 1;                          // tile_dim1 = 1
  g1[5] = (int)ndw;                   // tensor_dim0_stride[31:0]
  g1[6] = 0;
  g1[7] = 0;
  v4i32 z4; z4[0] = 0; z4[1] = 0; z4[2] = 0; z4[3] = 0;
  v8i32 z8;
#pragma unroll
  for (int i = 0; i < 8; ++i) z8[i] = 0;
  __builtin_amdgcn_tensor_load_to_lds(g0, g1, z4, z4, z8, 0);
#else
  const unsigned* s = (const unsigned*)gsrc;
  unsigned* d = (unsigned*)ldsDst;
  int lane = threadIdx.x & 31;
  for (unsigned i = lane; i < ndw; i += 32) d[i] = s[i];
#endif
}
__device__ __forceinline__ void tdm_wait0() {
#ifdef HAVE_TDM
  __builtin_amdgcn_s_wait_tensorcnt(0);
#endif
}
__device__ __forceinline__ void tdm_wait1() {
#ifdef HAVE_TDM
  __builtin_amdgcn_s_wait_tensorcnt(1);
#endif
}

// =====================================================================
// Kernel 1: xg0[t,b,0:256] = bf16WMMA( x[b,t,:] , W_ih0^T ) + (b_ih0+b_hh0)
// C = A[65536 x 832pad] * B[832 x 256], block tile 64x256 (8 waves, 8 acc)
// =====================================================================
__global__ __launch_bounds__(256) void k_xproj(
    const float* __restrict__ x, const float* __restrict__ Wih,
    const float* __restrict__ bih, const float* __restrict__ bhh,
    float* __restrict__ xg) {
  __shared__ __align__(32) __bf16 Af[4][32][16];     //  4 KB, fragment-major
  __shared__ __align__(32) __bf16 Bf[16][32][16];    // 16 KB, fragment-major
  __shared__ float biasS[256];

  const int tid  = threadIdx.x;
  const int wave = tid >> 5;
  const int lane = tid & 31;
  const int mtile = wave >> 1;           // 0..3
  const int ntBase = (wave & 1) * 8;     // 0 or 8
  const long mBase = (long)blockIdx.x * 64;

  biasS[tid] = bih[tid] + bhh[tid];

  v8f acc[8];
#pragma unroll
  for (int i = 0; i < 8; ++i) acc[i] = v8f_zero();

  for (int kc = 0; kc < KPAD; kc += 32) {
    if (kc + 32 <= IN_DIM) {
      // ---- full chunk: unpredicated staging ----
#pragma unroll
      for (int q = 0; q < 8; ++q) {       // A tile 64x32
        int idx = q * 256 + tid;
        int r = idx >> 5, k = idx & 31;
        float v = x[(mBase + r) * IN_DIM + (kc + k)];
        int fl, fe; a_frag_pos(r & 15, k, fl, fe);
        Af[r >> 4][fl][fe] = f2bf(v);
      }
#pragma unroll
      for (int q = 0; q < 32; ++q) {      // B tile 32x256: B[k][n] = Wih[n][k]
        int idx = q * 256 + tid;
        int n = idx >> 5, k = idx & 31;
        float v = Wih[n * IN_DIM + (kc + k)];
        int fl, fe; b_frag_pos(k, n & 15, fl, fe);
        Bf[n >> 4][fl][fe] = f2bf(v);
      }
    } else {
      // ---- tail chunk (kc=800): predicated, zero-padded ----
#pragma unroll
      for (int q = 0; q < 8; ++q) {
        int idx = q * 256 + tid;
        int r = idx >> 5, k = idx & 31;
        int kg = kc + k;
        float v = (kg < IN_DIM) ? x[(mBase + r) * IN_DIM + kg] : 0.0f;
        int fl, fe; a_frag_pos(r & 15, k, fl, fe);
        Af[r >> 4][fl][fe] = f2bf(v);
      }
#pragma unroll
      for (int q = 0; q < 32; ++q) {
        int idx = q * 256 + tid;
        int n = idx >> 5, k = idx & 31;
        int kg = kc + k;
        float v = (kg < IN_DIM) ? Wih[n * IN_DIM + kg] : 0.0f;
        int fl, fe; b_frag_pos(k, n & 15, fl, fe);
        Bf[n >> 4][fl][fe] = f2bf(v);
      }
    }
    if (kc + 32 < KPAD)  // global_prefetch_b8 of next K chunk of x
      __builtin_prefetch(&x[(mBase + (tid >> 2)) * IN_DIM + kc + 32], 0, 1);
    __syncthreads();

    v16bf a = *(const v16bf*)&Af[mtile][lane][0];
#pragma unroll
    for (int u = 0; u < 8; ++u) {
      v16bf b = *(const v16bf*)&Bf[ntBase + u][lane][0];
      acc[u] = __builtin_amdgcn_wmma_f32_16x16x32_bf16(
          false, a, false, b, (short)0, acc[u], false, false);
    }
    __syncthreads();
  }

  // ---- epilogue: bias add, store as xg[t, b, g] ----
  const int half = lane >> 4, colL = lane & 15;
#pragma unroll
  for (int u = 0; u < 8; ++u) {
#pragma unroll
    for (int e = 0; e < 8; ++e) {
      long gr = mBase + mtile * 16 + e + 8 * half;  // row in [B*T]
      int bb = (int)(gr >> 9);                      // b
      int tt = (int)(gr & 511);                     // t
      int col = (ntBase + u) * 16 + colL;
      xg[((long)tt * B_DIM + bb) * G_DIM + col] = acc[u][e] + biasS[col];
    }
  }
}

// =====================================================================
// Kernel 2: layer-0 LSTM scan. 8 WGs x 16 batch rows, persistent T loop.
// TDM double-buffers next step's xg tile (16KB contiguous) into LDS.
// =====================================================================
__global__ __launch_bounds__(256) void k_lstm0(
    const float* __restrict__ Whh, const float* __restrict__ xg,
    unsigned short* __restrict__ h1out) {
  __shared__ __align__(32) __bf16 WF[16][2][32][16];  // 32 KB B-fragments
  __shared__ __align__(32) __bf16 hA[2][32][16];      //  2 KB A-fragments (h)
  __shared__ float gates[16][256];                    // 16 KB
  __shared__ float cS[16][64];                        //  4 KB
  __shared__ __align__(16) float xgB[2][16 * 256];    // 32 KB TDM double buffer

  const int tid  = threadIdx.x;
  const int wave = tid >> 5;
  const int lane = tid & 31;
  const int br0  = blockIdx.x * 16;

  // prefetch xg tile for t=0 while we stage weights
  if (wave == 0)
    tdm_load_dwords(&xgB[0][0], xg + ((long)0 * B_DIM + br0) * G_DIM, 16 * 256);

  // stage W_hh0 as B fragments: B[k][n] = Whh[n*64 + k]
  for (int q = 0; q < 64; ++q) {
    int idx = q * 256 + tid;
    int n = idx & 255, k = idx >> 8;
    float v = Whh[n * 64 + k];
    int fl, fe; b_frag_pos(k & 31, n & 15, fl, fe);
    WF[n >> 4][k >> 5][fl][fe] = f2bf(v);
  }
#pragma unroll
  for (int q = 0; q < 4; ++q) {       // zero c and h
    int idx = q * 256 + tid;
    cS[idx >> 6][idx & 63] = 0.0f;
    ((__bf16*)hA)[idx] = f2bf(0.0f);
  }
  __syncthreads();

  const int half = lane >> 4, colL = lane & 15;

  for (int t = 0; t < T_DIM; ++t) {
    // issue prefetch of xg(t+1); guarantee xg(t) landed (in-order TDM)
    if (wave == 0) {
      if (t + 1 < T_DIM) {
        tdm_load_dwords(&xgB[(t + 1) & 1][0],
                        xg + ((long)(t + 1) * B_DIM + br0) * G_DIM, 16 * 256);
        tdm_wait1();
      } else {
        tdm_wait0();
      }
    }

    // ---- GEMM: G[16,256] = h[16,64] @ W[64,256]; wave -> 2 N tiles ----
#pragma unroll
    for (int u = 0; u < 2; ++u) {
      int nt = wave * 2 + u;
      v8f acc = v8f_zero();
#pragma unroll
      for (int ks = 0; ks < 2; ++ks) {
        v16bf a = *(const v16bf*)&hA[ks][lane][0];
        v16bf b = *(const v16bf*)&WF[nt][ks][lane][0];
        acc = __builtin_amdgcn_wmma_f32_16x16x32_bf16(
            false, a, false, b, (short)0, acc, false, false);
      }
#pragma unroll
      for (int e = 0; e < 8; ++e)
        gates[e + 8 * half][nt * 16 + colL] = acc[e];
    }
    __syncthreads();   // publishes gates AND the TDM'd xg(t) buffer

    // ---- elementwise gates (i,f,g,o) from LDS, state update ----
    const float* xr0 = &xgB[t & 1][0];
#pragma unroll
    for (int q = 0; q < 4; ++q) {
      int cell = q * 256 + tid;
      int m = cell >> 6, j = cell & 63;
      const float* xr = xr0 + m * G_DIM;
      float gi = gates[m][j]        + xr[j];
      float gf = gates[m][64 + j]   + xr[64 + j];
      float gg = gates[m][128 + j]  + xr[128 + j];
      float go = gates[m][192 + j]  + xr[192 + j];
      float iv = sigmoidf_(gi), fv = sigmoidf_(gf);
      float gv = tanhf(gg),     ov = sigmoidf_(go);
      float c = fv * cS[m][j] + iv * gv;
      float h = ov * tanhf(c);
      cS[m][j] = c;
      int fl, fe; a_frag_pos(m, j & 31, fl, fe);
      hA[j >> 5][fl][fe] = f2bf(h);
      h1out[((long)t * B_DIM + br0 + m) * H_DIM + j] = f2bf_bits(h);
    }
    __syncthreads();
  }
}

// =====================================================================
// Kernel 3: layer-1 LSTM scan, input projection fused:
// gates = [h1_t | h_prev](16x128) @ [W_ih1 | W_hh1]^T (128x256) + bias
// TDM double-buffers next step's h1 tile (2KB contiguous) into LDS.
// =====================================================================
__global__ __launch_bounds__(256) void k_lstm1(
    const float* __restrict__ Wih, const float* __restrict__ Whh,
    const float* __restrict__ bih, const float* __restrict__ bhh,
    const unsigned short* __restrict__ h1in,
    float* __restrict__ h2last) {
  __shared__ __align__(32) __bf16 WF[16][4][32][16];  // 64 KB B-fragments
  __shared__ __align__(32) __bf16 hA[4][32][16];      //  4 KB A-fragments
  __shared__ float gates[16][256];
  __shared__ float cS[16][64];
  __shared__ float biasS[256];
  __shared__ __align__(16) unsigned short h1B[2][16 * 64]; // 4 KB TDM buffer

  const int tid  = threadIdx.x;
  const int wave = tid >> 5;
  const int lane = tid & 31;
  const int br0  = blockIdx.x * 16;

  if (wave == 0)
    tdm_load_dwords(&h1B[0][0], h1in + ((long)0 * B_DIM + br0) * H_DIM,
                    (16 * 64) / 2);

  // stage combined weights: k<64 -> W_ih1, k>=64 -> W_hh1
  for (int q = 0; q < 128; ++q) {
    int idx = q * 256 + tid;
    int n = idx & 255, k = idx >> 8;   // k in 0..127
    float v = (k < 64) ? Wih[n * 64 + k] : Whh[n * 64 + (k - 64)];
    int fl, fe; b_frag_pos(k & 31, n & 15, fl, fe);
    WF[n >> 4][k >> 5][fl][fe] = f2bf(v);
  }
  biasS[tid] = bih[tid] + bhh[tid];
#pragma unroll
  for (int q = 0; q < 4; ++q) {
    int idx = q * 256 + tid;
    cS[idx >> 6][idx & 63] = 0.0f;
  }
#pragma unroll
  for (int q = 0; q < 4; ++q) {        // zero h_prev fragments (ksteps 2..3)
    int idx = q * 256 + tid;
    ((__bf16*)&hA[2][0][0])[idx] = f2bf(0.0f);
  }
  if (wave == 0) tdm_wait0();
  __syncthreads();

  // swizzle h1(0) into A-fragment ksteps 0..1
#pragma unroll
  for (int q = 0; q < 4; ++q) {
    int idx = q * 256 + tid;
    int m = idx >> 6, j = idx & 63;
    unsigned short raw = h1B[0][m * 64 + j];
    int fl, fe; a_frag_pos(m, j & 31, fl, fe);
    hA[j >> 5][fl][fe] = __builtin_bit_cast(__bf16, raw);
  }
  __syncthreads();

  const int half = lane >> 4, colL = lane & 15;

  for (int t = 0; t < T_DIM; ++t) {
    if (wave == 0 && t + 1 < T_DIM)
      tdm_load_dwords(&h1B[(t + 1) & 1][0],
                      h1in + ((long)(t + 1) * B_DIM + br0) * H_DIM,
                      (16 * 64) / 2);

    // ---- GEMM: G[16,256] = A[16,128] @ W[128,256] ----
#pragma unroll
    for (int u = 0; u < 2; ++u) {
      int nt = wave * 2 + u;
      v8f acc = v8f_zero();
#pragma unroll
      for (int ks = 0; ks < 4; ++ks) {
        v16bf a = *(const v16bf*)&hA[ks][lane][0];
        v16bf b = *(const v16bf*)&WF[nt][ks][lane][0];
        acc = __builtin_amdgcn_wmma_f32_16x16x32_bf16(
            false, a, false, b, (short)0, acc, false, false);
      }
#pragma unroll
      for (int e = 0; e < 8; ++e)
        gates[e + 8 * half][nt * 16 + colL] = acc[e];
    }
    __syncthreads();

    // ---- elementwise ----
#pragma unroll
    for (int q = 0; q < 4; ++q) {
      int cell = q * 256 + tid;
      int m = cell >> 6, j = cell & 63;
      float gi = gates[m][j]        + biasS[j];
      float gf = gates[m][64 + j]   + biasS[64 + j];
      float gg = gates[m][128 + j]  + biasS[128 + j];
      float go = gates[m][192 + j]  + biasS[192 + j];
      float iv = sigmoidf_(gi), fv = sigmoidf_(gf);
      float gv = tanhf(gg),     ov = sigmoidf_(go);
      float c = fv * cS[m][j] + iv * gv;
      float h = ov * tanhf(c);
      cS[m][j] = c;
      int fl, fe; a_frag_pos(m, j & 31, fl, fe);   // k = 64 + j
      hA[2 + (j >> 5)][fl][fe] = f2bf(h);
      if (t == T_DIM - 1) h2last[(br0 + m) * H_DIM + j] = h;
    }
    if (wave == 0) tdm_wait0();   // h1(t+1) landed (had GEMM+elementwise time)
    __syncthreads();

    // swizzle h1(t+1) into A-fragment ksteps 0..1 for next step
    if (t + 1 < T_DIM) {
#pragma unroll
      for (int q = 0; q < 4; ++q) {
        int idx = q * 256 + tid;
        int m = idx >> 6, j = idx & 63;
        unsigned short raw = h1B[(t + 1) & 1][m * 64 + j];
        int fl, fe; a_frag_pos(m, j & 31, fl, fe);
        hA[j >> 5][fl][fe] = __builtin_bit_cast(__bf16, raw);
      }
    }
    __syncthreads();
  }
}

// =====================================================================
// Kernel 4: out[b] = h2last[b,:] . fc_w + fc_b
// =====================================================================
__global__ __launch_bounds__(128) void k_fc(
    const float* __restrict__ h2, const float* __restrict__ fw,
    const float* __restrict__ fb, float* __restrict__ out) {
  int b = threadIdx.x;
  float s = fb[0];
#pragma unroll
  for (int j = 0; j < H_DIM; ++j) s += h2[b * H_DIM + j] * fw[j];
  out[b] = s;
}

// =====================================================================
extern "C" void kernel_launch(void* const* d_in, const int* in_sizes, int n_in,
                              void* d_out, int out_size, void* d_ws, size_t ws_size,
                              hipStream_t stream) {
  const float* x    = (const float*)d_in[0];
  const float* Wih0 = (const float*)d_in[1];
  const float* Whh0 = (const float*)d_in[2];
  const float* bih0 = (const float*)d_in[3];
  const float* bhh0 = (const float*)d_in[4];
  const float* Wih1 = (const float*)d_in[5];
  const float* Whh1 = (const float*)d_in[6];
  const float* bih1 = (const float*)d_in[7];
  const float* bhh1 = (const float*)d_in[8];
  const float* fcw  = (const float*)d_in[9];
  const float* fcb  = (const float*)d_in[10];

  char* ws = (char*)d_ws;
  size_t xg0_bytes = (size_t)T_DIM * B_DIM * G_DIM * sizeof(float);          // 64 MiB
  size_t h1_bytes  = (size_t)T_DIM * B_DIM * H_DIM * sizeof(unsigned short); //  8 MiB
  float*          xg0    = (float*)ws;
  unsigned short* h1     = (unsigned short*)(ws + xg0_bytes);
  float*          h2last = (float*)(ws + xg0_bytes + h1_bytes);

  k_xproj<<<(B_DIM * T_DIM) / 64, 256, 0, stream>>>(x, Wih0, bih0, bhh0, xg0);
  k_lstm0<<<B_DIM / 16, 256, 0, stream>>>(Whh0, xg0, h1);
  k_lstm1<<<B_DIM / 16, 256, 0, stream>>>(Wih1, Whh1, bih1, bhh1, h1, h2last);
  k_fc<<<1, 128, 0, stream>>>(h2last, fcw, fcb, (float*)d_out);
}
